// MultiheadAttention_44367012168192
// MI455X (gfx1250) — matrix operand order repaired and checked by reading, not compiled, not run on hardware
//
#include <hip/hip_runtime.h>
#include <hip/hip_bf16.h>

// ---------------------------------------------------------------------------
// MI455X (gfx1250) fused multi-head attention, softmax over HEAD axis.
// B=8, N=1024, D=768, H=12, Dh=64.  Heavy math on v_wmma_f32_16x16x32_bf16;
// weight tiles double-buffered into LDS via global_load_async_to_lds_b128
// (ASYNCcnt), overlapping the staging DMA with WMMA compute.
// ---------------------------------------------------------------------------

typedef __bf16 bf16_t;
typedef __attribute__((ext_vector_type(16))) __bf16 v16bf;
typedef __attribute__((ext_vector_type(8)))  __bf16 v8bf;
typedef __attribute__((ext_vector_type(8)))  float  v8f;

union BF16x16 {
    v16bf v;
    v8bf  h[2];
    unsigned int u[8];
};

// Build a 16-wide bf16 A/B fragment from two 16-byte chunks (ISA A/B layouts:
// each lane holds two 8-element runs of K, selected by lane half).
__device__ __forceinline__ v16bf ld_frag(const bf16_t* lo, const bf16_t* hi) {
    BF16x16 t;
    t.h[0] = *(const v8bf*)lo;
    t.h[1] = *(const v8bf*)hi;
    return t.v;
}

__device__ __forceinline__ unsigned int pk_bf16(float a, float b) {
    union { __bf16 h[2]; unsigned int u; } t;
    t.h[0] = (__bf16)a;
    t.h[1] = (__bf16)b;
    return t.u;
}

// ---------------------------------------------------------------------------
// fp32 -> bf16 conversion (x and the four weight matrices)
// ---------------------------------------------------------------------------
__global__ void cvt_f32_bf16(const float* __restrict__ in, bf16_t* __restrict__ out, int n) {
    int i = blockIdx.x * blockDim.x + threadIdx.x;
    if (i < n) out[i] = (bf16_t)in[i];
}

// ---------------------------------------------------------------------------
// Projection GEMM:  out = X @ W^T + bias,  X:[8192,768] bf16, W:[768,768] bf16.
// Block = 4 waves, 64x64 output tile.  Each wave owns a 16x64 strip
// (4 accumulators, one shared A-fragment per K=32 step).  The 64x32 W tile is
// DOUBLE-BUFFERED through LDS with async global->LDS copies: tile i+1 is in
// flight while tile i feeds the WMMAs (async loads complete in order, so
// s_wait_asynccnt 2 retires tile i and leaves tile i+1 outstanding).
// mode 0: -> Q  [B,H,N,Dh]         bf16, pre-scaled by 1/sqrt(Dh)
// mode 1: -> K2 [B,H][Dh/2][N][2]  bf16 (pair-packed along dh for B-frags)
// mode 2: -> V2 [B,H][N/2][Dh][2]  bf16 (pair-packed along m  for B-frags)
// mode 3: -> d_out [B,N,D]         f32  (final output projection)
// ---------------------------------------------------------------------------
__global__ __launch_bounds__(128) void proj_kernel(
    const bf16_t* __restrict__ X, const bf16_t* __restrict__ W,
    const float* __restrict__ bias, void* __restrict__ outp, int mode)
{
    __shared__ __align__(16) bf16_t tileB[2][64][32];   // double buffer, 8 KB

    const int lane  = threadIdx.x & 31;
    const int wave  = threadIdx.x >> 5;
    const int l16   = lane & 15;
    const int hiH   = (lane >= 16);
    const int m0    = blockIdx.x * 64 + wave * 16;
    const int n0    = blockIdx.y * 64;
    const int koffA = hiH ? 8  : 0;   // A: lane half -> K {0..7,16..23} vs {8..15,24..31}
    const int koffB = hiH ? 16 : 0;   // unused: B frags use koffA, same K order as A

    const bf16_t* xrow = X + (size_t)(m0 + l16) * 768;

    // Each thread async-copies two 16B chunks of the 64x32 W tile per k-step.
    const int ch0 = threadIdx.x * 2;          // chunk id 0..255
    const int bn0 = ch0 >> 2, bc0 = ch0 & 3;  // row in tile, 16B chunk in row
    const int bn1 = (ch0 + 1) >> 2, bc1 = (ch0 + 1) & 3;
    const bf16_t* wp0 = W + (size_t)(n0 + bn0) * 768 + bc0 * 8;
    const bf16_t* wp1 = W + (size_t)(n0 + bn1) * 768 + bc1 * 8;
    const unsigned lds0 = (unsigned)(uintptr_t)&tileB[0][bn0][bc0 * 8];
    const unsigned lds1 = (unsigned)(uintptr_t)&tileB[0][bn1][bc1 * 8];

    auto issue_tile = [&](int kk, int buf) {
        const unsigned boff = (unsigned)buf * (unsigned)sizeof(tileB[0]);
        asm volatile("global_load_async_to_lds_b128 %0, %1, off"
                     :: "v"(lds0 + boff), "v"(wp0 + kk) : "memory");
        asm volatile("global_load_async_to_lds_b128 %0, %1, off"
                     :: "v"(lds1 + boff), "v"(wp1 + kk) : "memory");
    };

    v8f acc[4] = {};

    issue_tile(0, 0);                         // prologue: tile 0 -> buf 0
    for (int kk = 0; kk < 768; kk += 32) {
        const int cur = (kk >> 5) & 1;
        if (kk + 32 < 768) {
            // Overlap: stage tile i+1 while tile i computes.  Safe: buf cur^1
            // was last read in iteration i-1, before that iteration's closing barrier.
            issue_tile(kk + 32, cur ^ 1);
            asm volatile("s_wait_asynccnt 0x2" ::: "memory");  // tile i landed
        } else {
            asm volatile("s_wait_asynccnt 0x0" ::: "memory");
        }
        __syncthreads();

        const v16bf a = ld_frag(xrow + kk + koffA, xrow + kk + 16 + koffA);
#pragma unroll
        for (int dt = 0; dt < 4; ++dt) {
            const bf16_t* brow = &tileB[cur][dt * 16 + l16][0];
            const v16bf bfr = ld_frag(brow + koffA, brow + 16 + koffA);
            acc[dt] = __builtin_amdgcn_wmma_f32_16x16x32_bf16(
                          false, a, false, bfr, (short)0, acc[dt], false, false);
        }
        __syncthreads();   // all waves done with buf cur before it is rewritten
    }

#pragma unroll
    for (int dt = 0; dt < 4; ++dt) {
        const int ncol = n0 + dt * 16 + l16;
        const float bn = bias[ncol];
        const int h  = ncol >> 6;
        const int dh = ncol & 63;
#pragma unroll
        for (int g = 0; g < 8; ++g) {
            const int r    = m0 + g + (hiH ? 8 : 0);   // C layout: VGPR g -> row g / g+8
            float v        = acc[dt][g] + bn;
            const int bIdx = r >> 10;
            const int nseq = r & 1023;
            if (mode == 3) {
                ((float*)outp)[(size_t)r * 768 + ncol] = v;
            } else {
                if (mode == 0) v *= 0.125f;   // fold 1/sqrt(Dh) into Q
                bf16_t* o = (bf16_t*)outp;
                const size_t base = (size_t)(bIdx * 12 + h) * (1024 * 64);
                if (mode == 0)
                    o[base + (size_t)nseq * 64 + dh] = (bf16_t)v;
                else if (mode == 1)
                    o[base + ((size_t)(dh >> 1) * 1024 + nseq) * 2 + (dh & 1)] = (bf16_t)v;
                else
                    o[base + ((size_t)(nseq >> 1) * 64 + dh) * 2 + (nseq & 1)] = (bf16_t)v;
            }
        }
    }
}

// ---------------------------------------------------------------------------
// Fused attention: scores (WMMA) -> head-axis softmax (LDS exchange) -> attn@V
// (WMMA).  One workgroup = one (b, 32-row n-tile); 12 waves = 12 heads, each
// wave processes two 16-row subtiles so K/V fragments are loaded once per
// m-tile and reused.  Scores never touch HBM (403 MB avoided).
// ---------------------------------------------------------------------------
__global__ __launch_bounds__(384) void attn_kernel(
    const bf16_t* __restrict__ Q, const bf16_t* __restrict__ K2,
    const bf16_t* __restrict__ V2, bf16_t* __restrict__ O)
{
    __shared__ float sS[12][32][32];   // per-head 32x32 score tiles (48 KB)

    const int lane = threadIdx.x & 31;
    const int h    = threadIdx.x >> 5;       // head = wave id
    const int l16  = lane & 15;
    const int hiH  = (lane >= 16);
    const int b    = blockIdx.x >> 5;        // 32 row-tiles per batch
    const int n0   = (blockIdx.x & 31) * 32;

    const size_t hbase = (size_t)(b * 12 + h) * (1024 * 64);
    const int koffA = hiH ? 8  : 0;
    const int koffB = hiH ? 16 : 0;

    // Q A-fragments for both 16-row subtiles (Q is pre-scaled by 1/sqrt(Dh))
    v16bf a0[2], a1[2];
#pragma unroll
    for (int st = 0; st < 2; ++st) {
        const bf16_t* qrow = Q + hbase + (size_t)(n0 + st * 16 + l16) * 64;
        a0[st] = ld_frag(qrow + koffA,      qrow + 16 + koffA);
        a1[st] = ld_frag(qrow + 32 + koffA, qrow + 48 + koffA);
    }

    const unsigned int* K2u = (const unsigned int*)(K2 + hbase); // [32][1024] u32
    const unsigned int* V2u = (const unsigned int*)(V2 + hbase); // [512][64]  u32

    v8f o[2][4] = {};

    for (int m0 = 0; m0 < 1024; m0 += 32) {
        // ---- K B-fragments, loaded once, reused by both subtiles -----------
        BF16x16 bk[2][2];                       // [m-half][k-step]
#pragma unroll
        for (int mh = 0; mh < 2; ++mh) {
            const int n = m0 + mh * 16 + l16;   // B column = key index m
#pragma unroll
            for (int g = 0; g < 8; ++g) {
                const int p = (g < 4) ? ((koffA >> 1) + g) : (8 + (koffA >> 1) + (g - 4));          // dh pair index
                bk[mh][0].u[g] = K2u[p * 1024 + n];      // dh 0..31
                bk[mh][1].u[g] = K2u[(p + 16) * 1024 + n];
            }
        }
        // ---- scores: two 16x32 subtiles ------------------------------------
#pragma unroll
        for (int st = 0; st < 2; ++st) {
            v8f s0 = {}, s1 = {};
            s0 = __builtin_amdgcn_wmma_f32_16x16x32_bf16(false, a0[st], false, bk[0][0].v, (short)0, s0, false, false);
            s0 = __builtin_amdgcn_wmma_f32_16x16x32_bf16(false, a1[st], false, bk[0][1].v, (short)0, s0, false, false);
            s1 = __builtin_amdgcn_wmma_f32_16x16x32_bf16(false, a0[st], false, bk[1][0].v, (short)0, s1, false, false);
            s1 = __builtin_amdgcn_wmma_f32_16x16x32_bf16(false, a1[st], false, bk[1][1].v, (short)0, s1, false, false);
#pragma unroll
            for (int g = 0; g < 8; ++g) {
                const int r = st * 16 + g + (hiH ? 8 : 0);
                sS[h][r][l16]      = s0[g];
                sS[h][r][16 + l16] = s1[g];
            }
        }
        __syncthreads();

        // ---- softmax across the 12 heads per (n,m) cell --------------------
        for (int cell = threadIdx.x; cell < 32 * 32; cell += 384) {
            const int r = cell >> 5, c = cell & 31;
            float mx = sS[0][r][c];
#pragma unroll
            for (int hh = 1; hh < 12; ++hh) mx = fmaxf(mx, sS[hh][r][c]);
            float e[12], sum = 0.f;
#pragma unroll
            for (int hh = 0; hh < 12; ++hh) { e[hh] = __expf(sS[hh][r][c] - mx); sum += e[hh]; }
            const float inv = 1.0f / sum;
#pragma unroll
            for (int hh = 0; hh < 12; ++hh) sS[hh][r][c] = e[hh] * inv;
        }
        __syncthreads();

        // ---- attn A-fragments (from LDS) + V B-fragments (reused) ----------
        BF16x16 aw[2];
#pragma unroll
        for (int st = 0; st < 2; ++st) {
            const int klo = hiH ? 8 : 0;
#pragma unroll
            for (int g = 0; g < 8; ++g) {
                const int k = (g < 4) ? (klo + 2 * g) : (16 + klo + 2 * (g - 4));
                aw[st].u[g] = pk_bf16(sS[h][st * 16 + l16][k], sS[h][st * 16 + l16][k + 1]);
            }
        }
#pragma unroll
        for (int dt = 0; dt < 4; ++dt) {
            BF16x16 bv;
            const int n = dt * 16 + l16;            // B column = dh
#pragma unroll
            for (int g = 0; g < 8; ++g) {
                const int p = (m0 >> 1) + ((g < 4) ? ((koffA >> 1) + g) : (8 + (koffA >> 1) + (g - 4)));  // m pair index
                bv.u[g] = V2u[p * 64 + n];
            }
#pragma unroll
            for (int st = 0; st < 2; ++st)
                o[st][dt] = __builtin_amdgcn_wmma_f32_16x16x32_bf16(
                                false, aw[st].v, false, bv.v, (short)0, o[st][dt], false, false);
        }
        // Safe LDS reuse: wave h is sole stash-writer / A-frag reader of
        // sS[h]; cross-head softmax traffic is bracketed by the two barriers.
        // No barrier between the attn@V phase and the next score phase, so the
        // scheduler may hoist next-tile K loads over the WMMAs.
    }

    // ---- store merged-head output O[B,N,768] bf16 --------------------------
#pragma unroll
    for (int st = 0; st < 2; ++st)
#pragma unroll
        for (int dt = 0; dt < 4; ++dt)
#pragma unroll
            for (int g = 0; g < 8; ++g) {
                const int r   = n0 + st * 16 + g + (hiH ? 8 : 0);
                const int col = h * 64 + dt * 16 + l16;
                O[((size_t)(b * 1024 + r)) * 768 + col] = (bf16_t)o[st][dt][g];
            }
}

// ---------------------------------------------------------------------------
extern "C" void kernel_launch(void* const* d_in, const int* in_sizes, int n_in,
                              void* d_out, int out_size, void* d_ws, size_t ws_size,
                              hipStream_t stream) {
    (void)in_sizes; (void)n_in; (void)out_size; (void)ws_size;

    const float* x  = (const float*)d_in[0];
    const float* Wq = (const float*)d_in[1];
    const float* bq = (const float*)d_in[2];
    const float* Wk = (const float*)d_in[3];
    const float* bk = (const float*)d_in[4];
    const float* Wv = (const float*)d_in[5];
    const float* bv = (const float*)d_in[6];
    const float* Wo = (const float*)d_in[7];
    const float* bo = (const float*)d_in[8];

    const size_t NX = (size_t)8 * 1024 * 768;  // 6,291,456
    const size_t NW = (size_t)768 * 768;       //   589,824

    bf16_t* xb  = (bf16_t*)d_ws;     // bf16 copies + intermediates (~68 MB)
    bf16_t* wqb = xb  + NX;
    bf16_t* wkb = wqb + NW;
    bf16_t* wvb = wkb + NW;
    bf16_t* wob = wvb + NW;
    bf16_t* Qb  = wob + NW;          // [B,H,N,Dh], pre-scaled
    bf16_t* K2  = Qb  + NX;          // [B,H][Dh/2][N][2]
    bf16_t* V2  = K2  + NX;          // [B,H][N/2][Dh][2]
    bf16_t* Ob  = V2  + NX;          // [B,N,D] pre-projection

    cvt_f32_bf16<<<(int)((NX + 255) / 256), 256, 0, stream>>>(x,  xb,  (int)NX);
    cvt_f32_bf16<<<(int)((NW + 255) / 256), 256, 0, stream>>>(Wq, wqb, (int)NW);
    cvt_f32_bf16<<<(int)((NW + 255) / 256), 256, 0, stream>>>(Wk, wkb, (int)NW);
    cvt_f32_bf16<<<(int)((NW + 255) / 256), 256, 0, stream>>>(Wv, wvb, (int)NW);
    cvt_f32_bf16<<<(int)((NW + 255) / 256), 256, 0, stream>>>(Wo, wob, (int)NW);

    const dim3 pgrid(128, 12);  // 8192/64 row tiles x 768/64 col tiles
    proj_kernel<<<pgrid, 128, 0, stream>>>(xb, wqb, bq, Qb, 0);
    proj_kernel<<<pgrid, 128, 0, stream>>>(xb, wkb, bk, K2, 1);
    proj_kernel<<<pgrid, 128, 0, stream>>>(xb, wvb, bv, V2, 2);

    attn_kernel<<<8 * 32, 384, 0, stream>>>(Qb, K2, V2, Ob);

    proj_kernel<<<pgrid, 128, 0, stream>>>(Ob, wob, bo, d_out, 3);
}
